// Transformer_17729624998372
// MI455X (gfx1250) — matrix operand order repaired and checked by reading, not compiled, not run on hardware
//
#include <hip/hip_runtime.h>

typedef __attribute__((ext_vector_type(16))) _Float16 v16h;
typedef __attribute__((ext_vector_type(2)))  _Float16 h2v;
typedef __attribute__((ext_vector_type(8)))  float    v8f;

// ===========================================================================
// f16-resident batched GEMM with async global->LDS staging.
//   C[z] = epilogue( A[z] @ B[z]^T )
//   A: f16 [M,K] row-major (lda, in halves), batch offset z1*sA1 + z2*sA2
//   B: f16 [N,K] row-major (ldb)   (all weights pre-transposed to [N,K])
//   C: f32 [M,N] (OUT16=0)  or f16 [M,N] (OUT16=1, TRANSC=0)
//      or f16 [N,M] (OUT16=1, TRANSC=1; ldc = leading stride of transposed C)
//   epilogue: +bias[n] (f32), optional ReLU, +R[m,n] (f32)
// Block tile 128x64, 256 threads = 8 wave32s; wave w owns rows w*16..w*16+15
// x all 64 cols -> 4 v_wmma_f32_16x16x32_f16 accumulators, A-frag reused.
// Double-buffered LDS filled by GLOBAL_LOAD_ASYNC_TO_LDS_B128 (ASYNCcnt),
// one barrier per K-step. Requires M%128==0, N%64==0, K%32==0,
// 16B alignment of all tile rows (holds for every GEMM in this model).
// ===========================================================================
#define A_LDS_STRIDE 40                    // halves per A/B LDS row (80B, 16B-mult)
#define A_BUF_HALVES (128 * A_LDS_STRIDE)  // one A buffer
#define B_BUF_HALVES (64 * A_LDS_STRIDE)   // one B buffer

__device__ __forceinline__ void async_ld16(unsigned lds_byte_addr, const _Float16* g)
{
    asm volatile("global_load_async_to_lds_b128 %0, %1, off"
                 :: "v"(lds_byte_addr),
                    "v"((unsigned long long)(__SIZE_TYPE__)g)
                 : "memory");
}

__device__ __forceinline__ void wait_async0()
{
    asm volatile("s_wait_asynccnt 0" ::: "memory");
}

template<int OUT16, int TRANSC>
__global__ __launch_bounds__(256)
void gemm_f16_async(const _Float16* __restrict__ A, long long sA1, long long sA2, int lda,
                    const _Float16* __restrict__ B, long long sB1, long long sB2, int ldb,
                    void* __restrict__ Cv, long long sC1, long long sC2, int ldc,
                    const float* __restrict__ bias, long long sb1, long long sb2,
                    const float* __restrict__ R, int ldr,
                    int M, int N, int K, int z2count, int relu)
{
    __shared__ _Float16 smem[2 * A_BUF_HALVES + 2 * B_BUF_HALVES];

    const int tid  = threadIdx.x;
    const int wave = tid >> 5;
    const int lane = tid & 31;
    const int lm   = lane & 15;
    const int lh   = lane >> 4;

    const int z  = blockIdx.z;
    const int z1 = z / z2count;
    const int z2 = z % z2count;

    const _Float16* Ab = A + (long long)z1 * sA1 + (long long)z2 * sA2;
    const _Float16* Bb = B + (long long)z1 * sB1 + (long long)z2 * sB2;
    const float* biasb = bias ? (bias + (long long)z1 * sb1 + (long long)z2 * sb2) : nullptr;

    const int n0 = blockIdx.x * 64;
    const int m0 = blockIdx.y * 128;

    v8f acc[4] = {};

    // LDS base byte address (flat LDS aperture: low 32 bits = LDS byte offset)
    const unsigned smem0 = (unsigned)(__SIZE_TYPE__)(&smem[0]);

    // async staging plan: A tile 128x32 halves = 512 16B-chunks -> 2/thread
    //                     B tile  64x32 halves = 256 16B-chunks -> 1/thread
    const int arow = tid >> 1;            // 0..127
    const int asub = (tid & 1) << 1;      // chunk pair base: 0 or 2 (8 halves each)
    const int brow = tid >> 2;            // 0..63
    const int bsub = tid & 3;             // 0..3

    auto issue = [&](int buf, int kt) {
        const _Float16* ga = Ab + (long long)(m0 + arow) * lda + kt + asub * 8;
        unsigned la = smem0 + (unsigned)((((buf * 128 + arow) * A_LDS_STRIDE) + asub * 8) * 2);
        async_ld16(la,      ga);
        async_ld16(la + 16, ga + 8);
        const _Float16* gb = Bb + (long long)(n0 + brow) * ldb + kt + bsub * 8;
        unsigned lb = smem0 + (unsigned)(((2 * A_BUF_HALVES) +
                         ((buf * 64 + brow) * A_LDS_STRIDE) + bsub * 8) * 2);
        async_ld16(lb, gb);
    };

    issue(0, 0);
    int cur = 0;
    for (int kt = 0; kt < K; kt += 32) {
        wait_async0();                      // my async copies into buf[cur] done
        __syncthreads();                    // publish (includes s_wait_dscnt 0)
        if (kt + 32 < K) issue(cur ^ 1, kt + 32);

        const _Float16* Asb = smem + cur * A_BUF_HALVES;
        const _Float16* Bsb = smem + 2 * A_BUF_HALVES + cur * B_BUF_HALVES;

        // ---- A fragment (ISA 7.12.2 16-bit A layout) ----
        v16h af;
        {
            const int am = wave * 16 + lm;
            #pragma unroll
            for (int v = 0; v < 8; ++v) {
                int kk = ((v >> 2) << 4) + ((v & 3) << 1) + (lh << 3);
                h2v pr = *reinterpret_cast<const h2v*>(&Asb[am * A_LDS_STRIDE + kk]);
                af[2 * v]     = pr[0];
                af[2 * v + 1] = pr[1];
            }
        }
        // ---- all 4 B fragments first, then 4 WMMAs ----
        v16h bf[4];
        #pragma unroll
        for (int s = 0; s < 4; ++s) {
            const int nb = s * 16 + lm;
            #pragma unroll
            for (int v = 0; v < 8; ++v) {
                int kk = ((v >> 2) << 4) + ((v & 3) << 1) + (lh << 3);
                h2v pr = *reinterpret_cast<const h2v*>(&Bsb[nb * A_LDS_STRIDE + kk]);
                bf[s][2 * v]     = pr[0];
                bf[s][2 * v + 1] = pr[1];
            }
        }
        #pragma unroll
        for (int s = 0; s < 4; ++s)
            acc[s] = __builtin_amdgcn_wmma_f32_16x16x32_f16(false, af, false, bf[s],
                                                            (short)0, acc[s], false, false);
        cur ^= 1;
    }

    // ---- epilogue ----
    float*    Cf = (float*)Cv;
    _Float16* Ch = (_Float16*)Cv;
    #pragma unroll
    for (int s = 0; s < 4; ++s) {
        int ng = n0 + s * 16 + lm;
        float bv = biasb ? biasb[ng] : 0.0f;
        #pragma unroll
        for (int v = 0; v < 8; ++v) {
            int mg = m0 + wave * 16 + v + (lh << 3);
            float val = acc[s][v] + bv;
            if (relu) val = fmaxf(val, 0.0f);
            if (R)    val += R[(long long)mg * ldr + ng];
            long long coff = ((long long)z1 * sC1 + (long long)z2 * sC2);
            if (OUT16) {
                long long idx = TRANSC ? ((long long)ng * ldc + mg)
                                       : ((long long)mg * ldc + ng);
                Ch[coff + idx] = (_Float16)val;
            } else {
                Cf[coff + (long long)mg * ldc + ng] = val;
            }
        }
    }
}

// ---------------------------------------------------------------------------
// Faithful multiplicative-mask softmax: masked[j] = (j>s) ? x[j]*(-1e6*scale) : 0
// reads f32 logits, writes f16 probabilities.
// ---------------------------------------------------------------------------
__global__ __launch_bounds__(256)
void softmax_mask_kernel(const float* __restrict__ logits, _Float16* __restrict__ probs,
                         float negscale)
{
    const int s = blockIdx.x;
    const long long z = blockIdx.y;
    const float* row = logits + (z * 512 + s) * 512ll;
    _Float16*    orow = probs + (z * 512 + s) * 512ll;
    const int tid = threadIdx.x;

    float r0 = row[tid];
    float r1 = row[tid + 256];
    float v0 = (tid > s)       ? r0 * negscale : 0.0f;
    float v1 = (tid + 256 > s) ? r1 * negscale : 0.0f;

    __shared__ float red[256];
    red[tid] = fmaxf(v0, v1);
    __syncthreads();
    for (int off = 128; off > 0; off >>= 1) {
        if (tid < off) red[tid] = fmaxf(red[tid], red[tid + off]);
        __syncthreads();
    }
    float rmax = red[0];
    __syncthreads();

    float e0 = __expf(v0 - rmax);
    float e1 = __expf(v1 - rmax);
    red[tid] = e0 + e1;
    __syncthreads();
    for (int off = 128; off > 0; off >>= 1) {
        if (tid < off) red[tid] += red[tid + off];
        __syncthreads();
    }
    float inv = 1.0f / red[0];
    orow[tid]       = (_Float16)(e0 * inv);
    orow[tid + 256] = (_Float16)(e1 * inv);
}

// ---------------------------------------------------------------------------
// LayerNorm over D=512: writes f32 (residual stream) and f16 (GEMM A operand)
// ---------------------------------------------------------------------------
__global__ __launch_bounds__(256)
void ln_kernel(const float* __restrict__ in, float* __restrict__ out,
               _Float16* __restrict__ outh,
               const float* __restrict__ g, const float* __restrict__ b)
{
    const long long row = blockIdx.x;
    const int tid = threadIdx.x;
    const float* x = in + row * 512;
    float v0 = x[tid];
    float v1 = x[tid + 256];

    __shared__ float s1[256];
    __shared__ float s2[256];
    s1[tid] = v0 + v1;
    s2[tid] = v0 * v0 + v1 * v1;
    __syncthreads();
    for (int off = 128; off > 0; off >>= 1) {
        if (tid < off) { s1[tid] += s1[tid + off]; s2[tid] += s2[tid + off]; }
        __syncthreads();
    }
    float mean = s1[0] * (1.0f / 512.0f);
    float var  = s2[0] * (1.0f / 512.0f) - mean * mean;
    float inv  = rsqrtf(var + 1e-5f);

    float y0 = (v0 - mean) * inv * g[tid]       + b[tid];
    float y1 = (v1 - mean) * inv * g[tid + 256] + b[tid + 256];
    out[row * 512 + tid]        = y0;
    out[row * 512 + tid + 256]  = y1;
    outh[row * 512 + tid]       = (_Float16)y0;
    outh[row * 512 + tid + 256] = (_Float16)y1;
}

// ---------------------------------------------------------------------------
// tokens(one-hot) @ (emb*sqrt(D)) + pos-encoding (exponent uses S=512);
// writes f32 + f16.
// ---------------------------------------------------------------------------
__global__ __launch_bounds__(256)
void embed_kernel(const float* __restrict__ tokens, const float* __restrict__ emb,
                  float* __restrict__ x, _Float16* __restrict__ xh)
{
    const int bs = blockIdx.x;          // b*512 + s
    const int s  = bs & 511;
    const int tid = threadIdx.x;
    const float* trow = tokens + (long long)bs * 1024;

    __shared__ int sid;
    if (tid == 0) sid = 0;
    __syncthreads();
    for (int j = tid; j < 1024; j += 256)
        if (trow[j] > 0.5f) sid = j;
    __syncthreads();

    const float* erow = emb + (long long)sid * 512;
    for (int d = tid; d < 512; d += 256) {
        float expo  = (2.0f * (float)d) * (1.0f / 512.0f);
        float denom = __powf(10000.0f, expo);
        float pe    = (float)s / denom;
        float pv    = (d & 1) ? __cosf(pe) : __sinf(pe);
        float val   = erow[d] * 22.62741699796952f + pv;
        x[(long long)bs * 512 + d]  = val;
        xh[(long long)bs * 512 + d] = (_Float16)val;
    }
}

// ---------------------------------------------------------------------------
// weight conversion: out[z][c][r] = (f16) in[z][r][c]   (fp32 -> f16 + transpose)
// ---------------------------------------------------------------------------
__global__ __launch_bounds__(256)
void tconv_kernel(const float* __restrict__ in, _Float16* __restrict__ out,
                  int rows, int cols, long long total)
{
    long long idx = (long long)blockIdx.x * 256 + threadIdx.x;
    if (idx >= total) return;
    long long rc = (long long)rows * cols;
    long long zz = idx / rc;
    long long rem = idx - zz * rc;
    int r = (int)(rem / cols);
    int c = (int)(rem - (long long)r * cols);
    out[zz * rc + (long long)c * rows + r] = (_Float16)in[idx];
}

__global__ __launch_bounds__(256)
void conv_kernel(const float* __restrict__ in, _Float16* __restrict__ out, long long n)
{
    long long idx = (long long)blockIdx.x * 256 + threadIdx.x;
    if (idx < n) out[idx] = (_Float16)in[idx];
}

// ---------------------------------------------------------------------------
extern "C" void kernel_launch(void* const* d_in, const int* in_sizes, int n_in,
                              void* d_out, int out_size, void* d_ws, size_t ws_size,
                              hipStream_t stream)
{
    (void)in_sizes; (void)n_in; (void)out_size; (void)ws_size;

    const float* tokens = (const float*)d_in[0];
    const float* emb    = (const float*)d_in[1];
    const float* qkv_w  = (const float*)d_in[2];
    const float* qkv_b  = (const float*)d_in[3];
    const float* out_w  = (const float*)d_in[4];
    const float* out_b  = (const float*)d_in[5];
    const float* w1     = (const float*)d_in[6];
    const float* b1     = (const float*)d_in[7];
    const float* w2     = (const float*)d_in[8];
    const float* b2     = (const float*)d_in[9];
    const float* ln1g   = (const float*)d_in[10];
    const float* ln1b   = (const float*)d_in[11];
    const float* ln2g   = (const float*)d_in[12];
    const float* ln2b   = (const float*)d_in[13];
    float* out = (float*)d_out;

    // ---------------- workspace layout ----------------
    float* wsf  = (float*)d_ws;
    float* xa   = wsf;                   // [4096,512] f32 residual stream
    float* xb   = xa + 2097152;          // [4096,512] f32 pre-LN temp
    float* abuf = xb + 2097152;          // [8,8,512,512] f32 logits (64MB)
    _Float16* wsh    = (_Float16*)(abuf + 16777216);
    _Float16* xah    = wsh;                      // [4096,512]
    _Float16* qkh    = xah    + 2097152;         // [H=8][4096][128] q|k
    _Float16* vTh    = qkh    + 4194304;         // [H=8][64][4096]  v transposed
    _Float16* attnh  = vTh    + 2097152;         // [8,8,512,512] f16 probs
    _Float16* ocath  = attnh  + 16777216;        // [4096,512]
    _Float16* qkv_wT = ocath  + 2097152;         // [L,H][192,512]
    _Float16* out_wT = qkv_wT + 4718592;         // [L][512,512]
    _Float16* w1T    = out_wT + 1572864;         // [L][2048,512]
    _Float16* w2T    = w1T    + 6291456;         // [L][512,2048]
    _Float16* embh   = w2T    + 6291456;         // [1024,512]
    _Float16* ff1h   = (_Float16*)abuf;          // [4096,2048] aliases logits region

    dim3 blk(256);

    // ---------------- one-time weight conversion (per call; deterministic) ---
    tconv_kernel<<<(4718592 + 255) / 256, blk, 0, stream>>>(qkv_w, qkv_wT, 512, 192, 4718592ll);
    tconv_kernel<<<(1572864 + 255) / 256, blk, 0, stream>>>(out_w, out_wT, 512, 512, 1572864ll);
    tconv_kernel<<<(6291456 + 255) / 256, blk, 0, stream>>>(w1,    w1T,    512, 2048, 6291456ll);
    tconv_kernel<<<(6291456 + 255) / 256, blk, 0, stream>>>(w2,    w2T,    2048, 512, 6291456ll);
    conv_kernel <<<(524288  + 255) / 256, blk, 0, stream>>>(emb,   embh,   524288ll);

    embed_kernel<<<4096, blk, 0, stream>>>(tokens, emb, xa, xah);

    for (int l = 0; l < 6; ++l) {
        const long long wqkvl = (long long)l * 8 * 192 * 512;

        // (1a) Q|K: xah[4096,512] @ qkv_wT[l,h][0:128,512]^T -> qkh[h][4096][128]
        gemm_f16_async<1, 0><<<dim3(2, 32, 8), blk, 0, stream>>>(
            xah, 0, 0, 512,
            qkv_wT + wqkvl, (long long)192 * 512, 0, 512,
            qkh, (long long)4096 * 128, 0, 128,
            qkv_b + (long long)l * 8 * 192, 192, 0,
            nullptr, 0,
            4096, 128, 512, 1, 0);

        // (1b) V (transposed C): -> vTh[h][64][4096]
        gemm_f16_async<1, 1><<<dim3(1, 32, 8), blk, 0, stream>>>(
            xah, 0, 0, 512,
            qkv_wT + wqkvl + (long long)128 * 512, (long long)192 * 512, 0, 512,
            vTh, (long long)64 * 4096, 0, 4096,
            qkv_b + (long long)l * 8 * 192 + 128, 192, 0,
            nullptr, 0,
            4096, 64, 512, 1, 0);

        // (2) logits[b,h] = Q @ K^T -> abuf f32   (z1=b, z2=h)
        gemm_f16_async<0, 0><<<dim3(8, 4, 64), blk, 0, stream>>>(
            qkh,      (long long)512 * 128, (long long)4096 * 128, 128,
            qkh + 64, (long long)512 * 128, (long long)4096 * 128, 128,
            abuf, (long long)8 * 512 * 512, (long long)512 * 512, 512,
            nullptr, 0, 0,
            nullptr, 0,
            512, 512, 64, 8, 0);

        // (3) multiplicative-mask softmax: f32 logits -> f16 probs
        softmax_mask_kernel<<<dim3(512, 64), blk, 0, stream>>>(abuf, attnh, -1.0e6f * 0.125f);

        // (4) o[b,h] = attn @ V -> ocath (head-concatenated, f16)
        gemm_f16_async<1, 0><<<dim3(1, 4, 64), blk, 0, stream>>>(
            attnh, (long long)8 * 512 * 512, (long long)512 * 512, 512,
            vTh,   512, (long long)64 * 4096, 4096,
            ocath, (long long)512 * 512, 64, 512,
            nullptr, 0, 0,
            nullptr, 0,
            512, 64, 512, 8, 0);

        // (5) out-proj + bias + residual(xa) -> xb (f32)
        gemm_f16_async<0, 0><<<dim3(8, 32, 1), blk, 0, stream>>>(
            ocath, 0, 0, 512,
            out_wT + (long long)l * 512 * 512, 0, 0, 512,
            xb, 0, 0, 512,
            out_b + (long long)l * 512, 0, 0,
            xa, 512,
            4096, 512, 512, 1, 0);

        // (6) LN1 -> xa (f32) + xah (f16)
        ln_kernel<<<4096, blk, 0, stream>>>(xb, xa, xah, ln1g + l * 512, ln1b + l * 512);

        // (7) FF1: relu(xah @ w1T^T + b1) -> ff1h (f16)
        gemm_f16_async<1, 0><<<dim3(32, 32, 1), blk, 0, stream>>>(
            xah, 0, 0, 512,
            w1T + (long long)l * 2048 * 512, 0, 0, 512,
            ff1h, 0, 0, 2048,
            b1 + (long long)l * 2048, 0, 0,
            nullptr, 0,
            4096, 2048, 512, 1, 1);

        // (8) FF2 + bias + residual(xa) -> xb (f32)
        gemm_f16_async<0, 0><<<dim3(8, 32, 1), blk, 0, stream>>>(
            ff1h, 0, 0, 2048,
            w2T + (long long)l * 512 * 2048, 0, 0, 2048,
            xb, 0, 0, 512,
            b2 + (long long)l * 512, 0, 0,
            xa, 512,
            4096, 512, 2048, 1, 0);

        // (9) LN2 -> xa + xah
        ln_kernel<<<4096, blk, 0, stream>>>(xb, xa, xah, ln2g + l * 512, ln2b + l * 512);
    }

    // final vocab projection: xah[4096,512] @ embh[1024,512]^T -> out f32
    gemm_f16_async<0, 0><<<dim3(16, 32, 1), blk, 0, stream>>>(
        xah, 0, 0, 512,
        embh, 0, 0, 512,
        out, 0, 0, 1024,
        nullptr, 0, 0,
        nullptr, 0,
        4096, 1024, 512, 1, 0);
}
